// DecoderLayer_39951785787468
// MI455X (gfx1250) — matrix operand order repaired and checked by reading, not compiled, not run on hardware
//
#include <hip/hip_runtime.h>
#include <hip/hip_bf16.h>

typedef __attribute__((ext_vector_type(16))) _Float16 v16h;
typedef __attribute__((ext_vector_type(8)))  float    v8f;

#define B_SZ   4
#define SQ_SZ  2048
#define SKV_SZ 4096
#define D_SZ   512

// ---------------------------------------------------------------------------
// Fragment loaders (wave32 WMMA 16x16x32 f16, this file's contiguous k order)
// A 16x32: lane m = l&15; lanes 0-15 khalf=0, lanes 16-31 khalf=16.
//   elements: k = khalf+0..15 -> a[0..15] (contig)
// B 32x16: lane n = l&15; lanes 0-15 k=0..15, lanes 16-31 k=16..31 (contig).
// ---------------------------------------------------------------------------
__device__ __forceinline__ v16h load_a_frag_f32(const float* __restrict__ base,
                                                int ld, int lane, int kbase) {
  const int m = lane & 15;
  const int khalf = (lane < 16) ? 0 : 16;
  const float* row = base + (size_t)m * ld + kbase + khalf;
  v16h a;
#pragma unroll
  for (int i = 0; i < 8; ++i) a[i] = (_Float16)row[i];
#pragma unroll
  for (int i = 0; i < 8; ++i) a[8 + i] = (_Float16)row[8 + i];
  return a;
}

__device__ __forceinline__ v16h load_a_frag_f16s(const _Float16* __restrict__ base,
                                                 int ld, int lane, float scale) {
  const int m = lane & 15;
  const int khalf = (lane < 16) ? 0 : 16;
  const _Float16* row = base + (size_t)m * ld + khalf;
  v16h a;
#pragma unroll
  for (int i = 0; i < 8; ++i) a[i] = (_Float16)((float)row[i] * scale);
#pragma unroll
  for (int i = 0; i < 8; ++i) a[8 + i] = (_Float16)((float)row[8 + i] * scale);
  return a;
}

__device__ __forceinline__ v16h load_b_frag_f32(const float* __restrict__ base,
                                                int ld, int lane, int kbase) {
  // B[k][n] = base[n*ld + k]  (column n of B is contiguous in memory)
  const int n = lane & 15;
  const int khalf = (lane < 16) ? 0 : 16;
  const float* col = base + (size_t)n * ld + kbase + khalf;
  v16h b;
#pragma unroll
  for (int i = 0; i < 16; ++i) b[i] = (_Float16)col[i];
  return b;
}

// ---------------------------------------------------------------------------
// Projection: Y = X @ W^T + bias, fp32 in -> f16 out.
// One 16x16 output tile per wave, 8 waves per 256-thread block.
// TRANSPOSE=true stores Yt[b, n, m] (used for V so PV B-frags are contiguous).
// ---------------------------------------------------------------------------
template <bool TRANSPOSE>
__global__ __launch_bounds__(256) void proj_kernel(
    const float* __restrict__ X,     // [B, M, D]
    const float* __restrict__ W,     // [D, D]
    const float* __restrict__ bias,  // [D]
    _Float16* __restrict__ Y,        // [B, M, D]  or  [B, D, M]
    int M) {
  const int lane = threadIdx.x & 31;
  const int wave = threadIdx.x >> 5;
  const int row0 = (blockIdx.x * 8 + wave) * 16;
  const int n0 = blockIdx.y * 16;
  const int b = blockIdx.z;

  const float* Xb = X + ((size_t)b * M + row0) * D_SZ;
  const float* Wn = W + (size_t)n0 * D_SZ;

  v8f acc = {};
#pragma unroll 4
  for (int k0 = 0; k0 < D_SZ; k0 += 32) {
    v16h a = load_a_frag_f32(Xb, D_SZ, lane, k0);
    v16h bf = load_b_frag_f32(Wn, D_SZ, lane, k0);
    acc = __builtin_amdgcn_wmma_f32_16x16x32_f16(false, a, false, bf, (short)0,
                                                 acc, false, false);
  }

  const int n = n0 + (lane & 15);
  const int mhi = (lane < 16) ? 0 : 8;
  const float bv = bias[n];
  if (!TRANSPOSE) {
    _Float16* out = Y + (size_t)b * M * D_SZ;
#pragma unroll
    for (int j = 0; j < 8; ++j) {
      const int m = row0 + j + mhi;
      out[(size_t)m * D_SZ + n] = (_Float16)(acc[j] + bv);
    }
  } else {
    _Float16* out = Y + (size_t)b * D_SZ * M;
#pragma unroll
    for (int j = 0; j < 8; ++j) {
      const int m = row0 + j + mhi;
      out[(size_t)n * M + m] = (_Float16)(acc[j] + bv);
    }
  }
}

// ---------------------------------------------------------------------------
// Flash attention over f16 Q/K/Vt with fp32 accumulation.
// Block = 4 waves (128 thr). Block owns 16 query rows of one batch.
// Each wave owns a 128-wide slice of D for both the QK^T reduction
// (partials summed through LDS) and the output accumulators.
// ---------------------------------------------------------------------------
__global__ __launch_bounds__(128) void attn_kernel(
    const _Float16* __restrict__ Qh,  // [B, SQ, D]
    const _Float16* __restrict__ Kh,  // [B, SKV, D]
    const _Float16* __restrict__ Vt,  // [B, D, SKV]
    float* __restrict__ Out) {        // [B, SQ, D]
  __shared__ float s_part[4][16][66];  // per-wave partial scores (padded)
  __shared__ float s_comb[16][68];     // combined 16x64 scores (padded)
  __shared__ float s_alpha[16];
  __shared__ float s_lsum[16];

  const int lane = threadIdx.x & 31;
  const int wave = threadIdx.x >> 5;
  const int hi = lane >> 4;   // half-wave select
  const int ln = lane & 15;
  const int b = blockIdx.y;
  const int q0 = blockIdx.x * 16;

  const float scale = 0.044194173824159216f;  // 1/sqrt(512), folded into Q

  // Q fragments for this wave's D-slice [wave*128, wave*128+128)
  const _Float16* qbase = Qh + ((size_t)b * SQ_SZ + q0) * D_SZ + wave * 128;
  v16h qfrag[4];
#pragma unroll
  for (int kf = 0; kf < 4; ++kf)
    qfrag[kf] = load_a_frag_f16s(qbase + kf * 32, D_SZ, lane, scale);

  v8f o[8];
#pragma unroll
  for (int t = 0; t < 8; ++t) o[t] = (v8f){};
  float m_i = -1e30f;
  float l_i = 0.0f;

  for (int t0 = 0; t0 < SKV_SZ; t0 += 64) {
    // Prefetch next K tile while we compute this one.
    if (t0 + 64 < SKV_SZ)
      __builtin_prefetch(
          Kh + ((size_t)b * SKV_SZ + t0 + 64 + ln) * D_SZ + wave * 128, 0, 1);

    // --- partial scores: S_w[16x64] = Qslice[16x128] @ Kslice^T -------------
#pragma unroll
    for (int g = 0; g < 4; ++g) {
      v8f s = {};
      const _Float16* kp =
          Kh + ((size_t)b * SKV_SZ + t0 + g * 16 + ln) * D_SZ + wave * 128 +
          hi * 16;
#pragma unroll
      for (int kf = 0; kf < 4; ++kf) {
        v16h bf = *(const v16h*)(kp + kf * 32);
        s = __builtin_amdgcn_wmma_f32_16x16x32_f16(false, qfrag[kf], false, bf,
                                                   (short)0, s, false, false);
      }
#pragma unroll
      for (int j = 0; j < 8; ++j)
        s_part[wave][j + hi * 8][g * 16 + ln] = s[j];
    }
    __syncthreads();

    // --- combine the 4 per-wave partials (128 thr x 8 elems) ---------------
    {
      const int tid = threadIdx.x;
#pragma unroll
      for (int e = 0; e < 8; ++e) {
        const int idx = tid * 8 + e;  // 0..1023
        const int r = idx >> 6, c = idx & 63;
        s_comb[r][c] = s_part[0][r][c] + s_part[1][r][c] + s_part[2][r][c] +
                       s_part[3][r][c];
      }
    }
    __syncthreads();

    // --- online softmax stats; each lane scans row ln (redundant x2/wave) --
    float rmax = -1e30f;
#pragma unroll 8
    for (int c = 0; c < 64; ++c) rmax = fmaxf(rmax, s_comb[ln][c]);
    const float m_new = fmaxf(m_i, rmax);
    float p_sum = 0.0f;
#pragma unroll 8
    for (int c = 0; c < 64; ++c) p_sum += __expf(s_comb[ln][c] - m_new);
    const float alpha = __expf(m_i - m_new);
    l_i = l_i * alpha + p_sum;
    m_i = m_new;
    if (wave == 0 && lane < 16) s_alpha[ln] = alpha;
    __syncthreads();

    // --- rescale accumulators (alpha is per output row) --------------------
    float aj[8];
#pragma unroll
    for (int j = 0; j < 8; ++j) aj[j] = s_alpha[j + hi * 8];
#pragma unroll
    for (int t = 0; t < 8; ++t)
#pragma unroll
      for (int j = 0; j < 8; ++j) o[t][j] *= aj[j];

    // --- O += P @ Vslice: P rebuilt from LDS into A-frag layout ------------
#pragma unroll
    for (int ks = 0; ks < 2; ++ks) {
      v16h p;
      const int kb = ks * 32 + hi * 16;
#pragma unroll
      for (int i = 0; i < 8; ++i)
        p[i] = (_Float16)__expf(s_comb[ln][kb + i] - m_i);
#pragma unroll
      for (int i = 0; i < 8; ++i)
        p[8 + i] = (_Float16)__expf(s_comb[ln][kb + 8 + i] - m_i);
#pragma unroll
      for (int t = 0; t < 8; ++t) {
        const int dout = wave * 128 + t * 16 + ln;
        const _Float16* vp =
            Vt + ((size_t)b * D_SZ + dout) * SKV_SZ + t0 + ks * 32 + hi * 16;
        v16h bf = *(const v16h*)vp;
        o[t] = __builtin_amdgcn_wmma_f32_16x16x32_f16(false, p, false, bf,
                                                      (short)0, o[t], false,
                                                      false);
      }
    }
    __syncthreads();
  }

  // --- normalize and store fp32 output -------------------------------------
  if (wave == 0 && lane < 16) s_lsum[ln] = l_i;
  __syncthreads();
  float inv[8];
#pragma unroll
  for (int j = 0; j < 8; ++j) inv[j] = 1.0f / s_lsum[j + hi * 8];

  float* outp = Out + ((size_t)b * SQ_SZ + q0) * D_SZ;
#pragma unroll
  for (int t = 0; t < 8; ++t) {
    const int n = wave * 128 + t * 16 + ln;
#pragma unroll
    for (int j = 0; j < 8; ++j) {
      const int m = j + hi * 8;
      outp[(size_t)m * D_SZ + n] = o[t][j] * inv[j];
    }
  }
}

// ---------------------------------------------------------------------------
extern "C" void kernel_launch(void* const* d_in, const int* in_sizes, int n_in,
                              void* d_out, int out_size, void* d_ws,
                              size_t ws_size, hipStream_t stream) {
  const float* x   = (const float*)d_in[0];  // [B, SQ, D]
  const float* enc = (const float*)d_in[1];  // [B, SKV, D]
  const float* wq  = (const float*)d_in[2];
  const float* bq  = (const float*)d_in[3];
  const float* wk  = (const float*)d_in[4];
  const float* bk  = (const float*)d_in[5];
  const float* wv  = (const float*)d_in[6];
  const float* bv  = (const float*)d_in[7];
  float* out = (float*)d_out;

  _Float16* Qh = (_Float16*)d_ws;                       // [B, SQ, D]
  _Float16* Kh = Qh + (size_t)B_SZ * SQ_SZ * D_SZ;      // [B, SKV, D]
  _Float16* Vt = Kh + (size_t)B_SZ * SKV_SZ * D_SZ;     // [B, D, SKV]

  // Projections: 8 waves/block, one 16x16 tile per wave.
  proj_kernel<false><<<dim3(SQ_SZ / 128, D_SZ / 16, B_SZ), 256, 0, stream>>>(
      x, wq, bq, Qh, SQ_SZ);
  proj_kernel<false><<<dim3(SKV_SZ / 128, D_SZ / 16, B_SZ), 256, 0, stream>>>(
      enc, wk, bk, Kh, SKV_SZ);
  proj_kernel<true><<<dim3(SKV_SZ / 128, D_SZ / 16, B_SZ), 256, 0, stream>>>(
      enc, wv, bv, Vt, SKV_SZ);

  // Attention: 16 query rows per block, 4 waves splitting D.
  attn_kernel<<<dim3(SQ_SZ / 16, B_SZ), 128, 0, stream>>>(Qh, Kh, Vt, out);
}
